// Decoder_71923522339343
// MI455X (gfx1250) — compile-verified, matrix operand order checked by reading, not run on hardware
//
#include <hip/hip_runtime.h>
#include <math.h>

#define VOCAB   50000
#define EMB     128
#define HID     512
#define TKLEN   400
#define BATCH   64
#define MAX_OOV 100
#define NEXT    (VOCAB + MAX_OOV)   // 50100
#define N2H     (2 * HID)           // 1024
#define NROWS   (BATCH * TKLEN)     // 25600

typedef __attribute__((ext_vector_type(16))) __bf16 v16bf;
typedef __attribute__((ext_vector_type(8)))  __bf16 v8bf;
typedef __attribute__((ext_vector_type(8)))  float  v8f;

// branch-free transcendentals on the hardware v_exp_f32 / v_rcp_f32 path
__device__ __forceinline__ float fast_tanh(float x) {
    float e = __expf(2.f * x);                        // inf for large x, 0 for -large
    return 1.f - 2.f * __builtin_amdgcn_rcpf(e + 1.f); // -> +/-1 at the limits
}
__device__ __forceinline__ float fast_sigmoid(float x) {
    return __builtin_amdgcn_rcpf(1.f + __expf(-x));
}

// ---------------------------------------------------------------------------
// 0. zero helper (scores2 is accumulated with global atomics)
// ---------------------------------------------------------------------------
__global__ void k_zero(float* __restrict__ p, int n) {
    int i = blockIdx.x * blockDim.x + threadIdx.x;
    if (i < n) p[i] = 0.f;
}

// ---------------------------------------------------------------------------
// 1. x = Linear([c_t_1, emb_W[y]], xc_W, xc_b)   (64 x 128)
// ---------------------------------------------------------------------------
__global__ void k_embed_x(const int* __restrict__ y, const float* __restrict__ emb_W,
                          const float* __restrict__ c_t_1, const float* __restrict__ xc_W,
                          const float* __restrict__ xc_b, float* __restrict__ x_out) {
    int b = blockIdx.x, e = threadIdx.x;
    const float* w  = xc_W + (size_t)e * (N2H + EMB);
    const float* c  = c_t_1 + (size_t)b * N2H;
    const float* ye = emb_W + (size_t)y[b] * EMB;
    float acc = xc_b[e];
    for (int j = 0; j < N2H; ++j) acc += w[j] * c[j];
    for (int j = 0; j < EMB; ++j) acc += w[N2H + j] * ye[j];
    x_out[b * EMB + e] = acc;
}

// ---------------------------------------------------------------------------
// 2. single-step LSTM (torch gate order i,f,g,o) -> h_new, c_new, s_t_hat
// ---------------------------------------------------------------------------
__global__ void k_lstm(const float* __restrict__ x, const float* __restrict__ s_h,
                       const float* __restrict__ s_c, const float* __restrict__ W_ih,
                       const float* __restrict__ W_hh, const float* __restrict__ b_ih,
                       const float* __restrict__ b_hh, float* __restrict__ s_t_hat,
                       float* __restrict__ h_out, float* __restrict__ c_out) {
    int b = blockIdx.x, h = threadIdx.x;
    const float* xb = x + b * EMB;
    const float* h0 = s_h + b * HID;
    float g[4];
    #pragma unroll
    for (int gi = 0; gi < 4; ++gi) {
        int j = gi * HID + h;
        float acc = b_ih[j] + b_hh[j];
        const float* wi = W_ih + (size_t)j * EMB;
        for (int e = 0; e < EMB; ++e) acc += wi[e] * xb[e];
        const float* wh = W_hh + (size_t)j * HID;
        for (int k = 0; k < HID; ++k) acc += wh[k] * h0[k];
        g[gi] = acc;
    }
    float ig = fast_sigmoid(g[0]);
    float fg = fast_sigmoid(g[1]);
    float gg = fast_tanh(g[2]);
    float og = fast_sigmoid(g[3]);
    float c0 = s_c[b * HID + h];
    float cn = fg * c0 + ig * gg;
    float hn = og * fast_tanh(cn);
    h_out[b * HID + h] = hn;
    c_out[b * HID + h] = cn;
    s_t_hat[b * N2H + h] = hn;
    s_t_hat[b * N2H + HID + h] = cn;
}

// ---------------------------------------------------------------------------
// 3. dec_fea = s_t_hat @ Ws^T + b   (64 x 1024)
// ---------------------------------------------------------------------------
__global__ void k_decfea(const float* __restrict__ s_t_hat, const float* __restrict__ Ws_W,
                         const float* __restrict__ Ws_b, float* __restrict__ dec_fea) {
    int b = blockIdx.x, n = threadIdx.x;              // blockDim = 1024
    const float* w = Ws_W + (size_t)n * N2H;
    const float* s = s_t_hat + b * N2H;
    float acc = Ws_b[n];
    for (int k = 0; k < N2H; ++k) acc += w[k] * s[k];
    dec_fea[b * N2H + n] = acc;
}

// ---------------------------------------------------------------------------
// 4. coverage attention scores: memory bound over encoder_feature (105 MB);
//    branch-free tanh keeps the 26.2M-element pass on the bandwidth roofline.
// ---------------------------------------------------------------------------
__global__ void __launch_bounds__(256)
k_cov_scores(const float* __restrict__ enc_feat, const float* __restrict__ dec_fea,
             const float* __restrict__ coverage, const float* __restrict__ Wc,
             const float* __restrict__ vv, float* __restrict__ scores) {
    int tid = threadIdx.x;
    int wave = tid >> 5, lane = tid & 31;
    int row = blockIdx.x * 8 + wave;                  // < 25600
    int b = row / TKLEN;
    float cov = coverage[row];
    const float* ef = enc_feat + (size_t)row * N2H;
    const float* df = dec_fea + (size_t)b * N2H;
    float acc = 0.f;
    for (int n = lane; n < N2H; n += 32)
        acc += fast_tanh(ef[n] + df[n] + cov * Wc[n]) * vv[n];
    for (int off = 16; off > 0; off >>= 1) acc += __shfl_down(acc, off, 32);
    if (lane == 0) scores[row] = acc;
}

// ---------------------------------------------------------------------------
// 5. structural attention: WMMA bf16 GEMM (25600x512 @ 512x1024) fused with
//    +dec_fea -> tanh -> * v[n] -> reduce over n  => scores2 (pre-zeroed).
//    64 rows/block (4 row tiles): each Wr fragment feeds 4 WMMAs; Wr L2
//    traffic = 2MB * 400 blocks = 800 MB. LDS A tile is bf16 with per-row
//    8-element rotation -> conflict-free ds_load_b128 fragments.
// ---------------------------------------------------------------------------
__global__ void __launch_bounds__(256)
k_struct_scores_wmma(const float* __restrict__ ri, const float* __restrict__ Wr,
                     const float* __restrict__ dec_fea, const float* __restrict__ vv,
                     float* __restrict__ scores2) {
    __shared__ __align__(16) __bf16 Asm[64][512];     // 64 KB, row-swizzled
    const int tid  = threadIdx.x;
    const int wave = tid >> 5, lane = tid & 31;
    const int row0 = blockIdx.x * 64;

    // stage A tile (ri rows), f32 -> bf16, swizzle row m by m*8 elements
    for (int i = tid; i < 64 * HID; i += 256) {
        int m = i >> 9, k = i & 511;
        Asm[m][(k + m * 8) & 511] = (__bf16)ri[(size_t)(row0 + m) * HID + k];
    }
    __syncthreads();

    const int khalf = lane >> 4;                      // CDNA5 16-bit frag K half
    const int nlane = lane & 15;
    float priv[4][8];
    #pragma unroll
    for (int mt = 0; mt < 4; ++mt)
        #pragma unroll
        for (int r = 0; r < 8; ++r) priv[mt][r] = 0.f;

    for (int ct = 0; ct < 8; ++ct) {                  // 8 column tiles per wave
        const int n = wave * 128 + ct * 16 + nlane;
        v8f acc[4] = {};
        for (int k0 = 0; k0 < HID; k0 += 32) {
            // B fragment from Wr (row n, CDNA5 K packing), f32 -> bf16
            const float* wr = Wr + (size_t)n * HID + k0 + khalf * 8;
            v16bf bm;
            #pragma unroll
            for (int j = 0; j < 8; ++j) bm[j] = (__bf16)wr[j];
            #pragma unroll
            for (int j = 0; j < 8; ++j) bm[j + 8] = (__bf16)wr[16 + j];
            #pragma unroll
            for (int mt = 0; mt < 4; ++mt) {
                int m  = mt * 16 + nlane;
                int ka = (k0 + khalf * 8 + m * 8) & 511;
                int kb = (k0 + 16 + khalf * 8 + m * 8) & 511;
                v8bf lo = *(const v8bf*)&Asm[m][ka];
                v8bf hi = *(const v8bf*)&Asm[m][kb];
                v16bf a;
                #pragma unroll
                for (int j = 0; j < 8; ++j) { a[j] = lo[j]; a[j + 8] = hi[j]; }
                acc[mt] = __builtin_amdgcn_wmma_f32_16x16x32_bf16(
                    false, a, false, bm, (short)0, acc[mt], false, false);
            }
        }
        const float vn = vv[n];
        #pragma unroll
        for (int mt = 0; mt < 4; ++mt)
            #pragma unroll
            for (int r = 0; r < 8; ++r) {             // C layout: M=r(+8), N=lane&15
                int m = mt * 16 + ((lane < 16) ? r : (r + 8));
                int bb = (row0 + m) / TKLEN;
                priv[mt][r] += fast_tanh(acc[mt][r] + dec_fea[(size_t)bb * N2H + n]) * vn;
            }
    }
    // reduce over the 16 lanes sharing a row, then one atomic per (row, wave)
    #pragma unroll
    for (int mt = 0; mt < 4; ++mt)
        #pragma unroll
        for (int r = 0; r < 8; ++r) {
            float val = priv[mt][r];
            val += __shfl_xor(val, 8, 32);
            val += __shfl_xor(val, 4, 32);
            val += __shfl_xor(val, 2, 32);
            val += __shfl_xor(val, 1, 32);
            if (nlane == 0) {
                int m = mt * 16 + ((lane < 16) ? r : (r + 8));
                atomicAdd(&scores2[row0 + m], val);
            }
        }
}

// ---------------------------------------------------------------------------
// 6. softmax over t (masked, optional renorm) + context c = a @ enc_out
// ---------------------------------------------------------------------------
__global__ void __launch_bounds__(256)
k_softmax_ctx(const float* __restrict__ scores, const float* __restrict__ mask,
              const float* __restrict__ cov_in, const float* __restrict__ enc_out,
              float* __restrict__ a_out, float* __restrict__ ctx_out,
              float* attn_copy, float* cov_out, int renorm) {
    __shared__ float red[256];
    __shared__ float a_sh[TKLEN];
    int b = blockIdx.x, tid = threadIdx.x;
    const float* sc = scores + b * TKLEN;
    const float* mk = mask + b * TKLEN;
    float m = -1e30f;
    for (int t = tid; t < TKLEN; t += 256) m = fmaxf(m, sc[t]);
    red[tid] = m; __syncthreads();
    for (int s = 128; s > 0; s >>= 1) { if (tid < s) red[tid] = fmaxf(red[tid], red[tid + s]); __syncthreads(); }
    float mx = red[0]; __syncthreads();
    float se = 0.f, sm = 0.f;
    for (int t = tid; t < TKLEN; t += 256) {
        float e = __expf(sc[t] - mx);
        se += e; sm += e * mk[t];
    }
    red[tid] = se; __syncthreads();
    for (int s = 128; s > 0; s >>= 1) { if (tid < s) red[tid] += red[tid + s]; __syncthreads(); }
    float sumall = red[0]; __syncthreads();
    red[tid] = sm; __syncthreads();
    for (int s = 128; s > 0; s >>= 1) { if (tid < s) red[tid] += red[tid + s]; __syncthreads(); }
    float summask = red[0]; __syncthreads();
    float dinv = __builtin_amdgcn_rcpf(renorm ? summask : sumall);
    for (int t = tid; t < TKLEN; t += 256) {
        float a = __expf(sc[t] - mx) * mk[t] * dinv;
        a_sh[t] = a;
        a_out[b * TKLEN + t] = a;
        if (attn_copy) attn_copy[b * TKLEN + t] = a;
        if (cov_out)   cov_out[b * TKLEN + t] = cov_in[b * TKLEN + t] + a;
    }
    __syncthreads();
    const float* eb = enc_out + (size_t)b * TKLEN * N2H;
    for (int n = tid; n < N2H; n += 256) {
        float acc = 0.f;
        for (int t = 0; t < TKLEN; ++t) acc += a_sh[t] * eb[(size_t)t * N2H + n];
        ctx_out[b * N2H + n] = acc;
    }
}

// ---------------------------------------------------------------------------
// 7. p_gen = sigmoid(pg_W . [c_struct, s_t_hat, x] + pg_b)
// ---------------------------------------------------------------------------
__global__ void k_pgen(const float* __restrict__ c_struct, const float* __restrict__ s_t_hat,
                       const float* __restrict__ x, const float* __restrict__ pg_W,
                       const float* __restrict__ pg_b, float* __restrict__ pgen,
                       float* __restrict__ pgen_out) {
    int b = blockIdx.x, lane = threadIdx.x;
    float acc = 0.f;
    for (int j = lane; j < N2H; j += 32) acc += pg_W[j]           * c_struct[b * N2H + j];
    for (int j = lane; j < N2H; j += 32) acc += pg_W[N2H + j]     * s_t_hat[b * N2H + j];
    for (int j = lane; j < EMB; j += 32) acc += pg_W[2 * N2H + j] * x[b * EMB + j];
    for (int off = 16; off > 0; off >>= 1) acc += __shfl_down(acc, off, 32);
    if (lane == 0) {
        float p = fast_sigmoid(acc + pg_b[0]);
        pgen[b] = p; pgen_out[b] = p;
    }
}

// ---------------------------------------------------------------------------
// 8. out1 = Linear([h_new, c_struct], o1)   (64 x 512)
// ---------------------------------------------------------------------------
__global__ void k_out1(const float* __restrict__ s_t_hat, const float* __restrict__ c_struct,
                       const float* __restrict__ o1_W, const float* __restrict__ o1_b,
                       float* __restrict__ out1) {
    int b = blockIdx.x, h = threadIdx.x;
    const float* w  = o1_W + (size_t)h * (3 * HID);
    const float* hn = s_t_hat + b * N2H;              // first HID = h_new
    const float* cs = c_struct + b * N2H;
    float acc = o1_b[h];
    for (int j = 0; j < HID; ++j) acc += w[j] * hn[j];
    for (int j = 0; j < N2H; ++j) acc += w[HID + j] * cs[j];
    out1[b * HID + h] = acc;
}

// ---------------------------------------------------------------------------
// 9. logits = out1 @ o2_W^T + o2_b : WMMA bf16, all 64 rows in one block so
//    the dominant 102 MB o2_W stream is read exactly once from HBM.
// ---------------------------------------------------------------------------
__global__ void __launch_bounds__(256)
k_logits_wmma(const float* __restrict__ out1, const float* __restrict__ o2_W,
              const float* __restrict__ o2_b, float* __restrict__ logits) {
    __shared__ __align__(16) __bf16 Asm[64][512];     // 64 KB, row-swizzled
    const int tid  = threadIdx.x;
    const int wave = tid >> 5, lane = tid & 31;
    const int n0 = blockIdx.x * 128 + wave * 16;
    for (int i = tid; i < 64 * HID; i += 256) {
        int m = i >> 9, k = i & 511;
        Asm[m][(k + m * 8) & 511] = (__bf16)out1[(size_t)m * HID + k];
    }
    __syncthreads();
    const int khalf = lane >> 4;
    const int nlane = lane & 15;
    const int n = n0 + nlane;
    const bool nvalid = (n < VOCAB);
    const float* wn = o2_W + (size_t)(nvalid ? n : 0) * HID;
    v8f acc[4] = {};
    for (int k0 = 0; k0 < HID; k0 += 32) {
        const float* wk = wn + k0 + khalf * 8;
        if (k0 + 32 < HID) __builtin_prefetch(wk + 32, 0, 0);
        v16bf bm;
        #pragma unroll
        for (int j = 0; j < 8; ++j) bm[j] = (__bf16)wk[j];
        #pragma unroll
        for (int j = 0; j < 8; ++j) bm[j + 8] = (__bf16)wk[16 + j];
        #pragma unroll
        for (int mt = 0; mt < 4; ++mt) {
            int m  = mt * 16 + nlane;
            int ka = (k0 + khalf * 8 + m * 8) & 511;
            int kb = (k0 + 16 + khalf * 8 + m * 8) & 511;
            v8bf lo = *(const v8bf*)&Asm[m][ka];
            v8bf hi = *(const v8bf*)&Asm[m][kb];
            v16bf a;
            #pragma unroll
            for (int j = 0; j < 8; ++j) { a[j] = lo[j]; a[j + 8] = hi[j]; }
            acc[mt] = __builtin_amdgcn_wmma_f32_16x16x32_bf16(
                false, a, false, bm, (short)0, acc[mt], false, false);
        }
    }
    if (nvalid) {
        const float bias = o2_b[n];
        #pragma unroll
        for (int mt = 0; mt < 4; ++mt)
            #pragma unroll
            for (int r = 0; r < 8; ++r) {
                int m = mt * 16 + ((lane < 16) ? r : (r + 8));
                logits[(size_t)m * VOCAB + n] = acc[mt][r] + bias;
            }
    }
}

// ---------------------------------------------------------------------------
// 10. vocab softmax * p_gen -> final_dist base, zero OOV slots
// ---------------------------------------------------------------------------
__global__ void __launch_bounds__(256)
k_vocab_softmax(const float* __restrict__ logits, const float* __restrict__ pgen,
                float* __restrict__ final_dist) {
    __shared__ float red[256];
    int b = blockIdx.x, tid = threadIdx.x;
    const float* lg = logits + (size_t)b * VOCAB;
    float m = -1e30f;
    for (int v = tid; v < VOCAB; v += 256) m = fmaxf(m, lg[v]);
    red[tid] = m; __syncthreads();
    for (int s = 128; s > 0; s >>= 1) { if (tid < s) red[tid] = fmaxf(red[tid], red[tid + s]); __syncthreads(); }
    float mx = red[0]; __syncthreads();
    float se = 0.f;
    for (int v = tid; v < VOCAB; v += 256) se += __expf(lg[v] - mx);
    red[tid] = se; __syncthreads();
    for (int s = 128; s > 0; s >>= 1) { if (tid < s) red[tid] += red[tid + s]; __syncthreads(); }
    float inv = __builtin_amdgcn_rcpf(red[0]);
    float p = pgen[b];
    float* fd = final_dist + (size_t)b * NEXT;
    for (int v = tid; v < VOCAB; v += 256) fd[v] = p * __expf(lg[v] - mx) * inv;
    for (int v = VOCAB + tid; v < NEXT; v += 256) fd[v] = 0.f;
}

// ---------------------------------------------------------------------------
// 11. pointer scatter: final[b, ext_idx] += (1-p_gen) * attn
// ---------------------------------------------------------------------------
__global__ void k_scatter(const int* __restrict__ ext_idx, const float* __restrict__ attn,
                          const float* __restrict__ pgen, float* __restrict__ final_dist) {
    int i = blockIdx.x * blockDim.x + threadIdx.x;
    if (i >= NROWS) return;
    int b = i / TKLEN;
    atomicAdd(&final_dist[(size_t)b * NEXT + ext_idx[i]], (1.f - pgen[b]) * attn[i]);
}

// ---------------------------------------------------------------------------
extern "C" void kernel_launch(void* const* d_in, const int* in_sizes, int n_in,
                              void* d_out, int out_size, void* d_ws, size_t ws_size,
                              hipStream_t stream) {
    const int*   y_t_1    = (const int*)  d_in[0];
    const float* s_h      = (const float*)d_in[1];
    const float* s_c      = (const float*)d_in[2];
    const float* enc_out  = (const float*)d_in[3];
    const float* enc_feat = (const float*)d_in[4];
    const float* mask     = (const float*)d_in[5];
    const float* c_t_1    = (const float*)d_in[6];
    const int*   ext_voc  = (const int*)  d_in[8];
    const float* coverage = (const float*)d_in[9];
    const float* ri       = (const float*)d_in[11];
    const float* emb_W    = (const float*)d_in[12];
    const float* Ws_W     = (const float*)d_in[13];
    const float* Ws_b     = (const float*)d_in[14];
    const float* v_W      = (const float*)d_in[15];
    const float* Wc_W     = (const float*)d_in[16];
    const float* Wr_W     = (const float*)d_in[17];
    const float* xc_W     = (const float*)d_in[18];
    const float* xc_b     = (const float*)d_in[19];
    const float* W_ih     = (const float*)d_in[20];
    const float* W_hh     = (const float*)d_in[21];
    const float* b_ih     = (const float*)d_in[22];
    const float* b_hh     = (const float*)d_in[23];
    const float* pg_W     = (const float*)d_in[24];
    const float* pg_b     = (const float*)d_in[25];
    const float* o1_W     = (const float*)d_in[26];
    const float* o1_b     = (const float*)d_in[27];
    const float* o2_W     = (const float*)d_in[28];
    const float* o2_b     = (const float*)d_in[29];

    float* out = (float*)d_out;
    // output layout (flat, return order)
    const size_t OF_FINAL = 0;
    const size_t OF_H    = OF_FINAL + (size_t)BATCH * NEXT;     // 3,206,400
    const size_t OF_C    = OF_H + (size_t)BATCH * HID;
    const size_t OF_CT   = OF_C + (size_t)BATCH * HID;
    const size_t OF_ATTN = OF_CT + (size_t)BATCH * N2H;
    const size_t OF_PGEN = OF_ATTN + (size_t)BATCH * TKLEN;
    const size_t OF_COV  = OF_PGEN + (size_t)BATCH;

    float* ws = (float*)d_ws;
    float* w_x    = ws;                                   // 64*128
    float* w_sth  = w_x    + (size_t)BATCH * EMB;         // 64*1024
    float* w_dec  = w_sth  + (size_t)BATCH * N2H;         // 64*1024
    float* w_sc1  = w_dec  + (size_t)BATCH * N2H;         // 25600
    float* w_sc2  = w_sc1  + NROWS;                       // 25600 (atomic accum)
    float* w_att  = w_sc2  + NROWS;                       // 25600
    float* w_ast  = w_att  + NROWS;                       // 25600
    float* w_cst  = w_ast  + NROWS;                       // 64*1024
    float* w_o1   = w_cst  + (size_t)BATCH * N2H;         // 64*512
    float* w_pg   = w_o1   + (size_t)BATCH * HID;         // 64
    float* w_log  = w_pg   + BATCH;                       // 64*50000 (12.8 MB)

    k_zero<<<(NROWS + 255) / 256, 256, 0, stream>>>(w_sc2, NROWS);
    k_embed_x<<<BATCH, EMB, 0, stream>>>(y_t_1, emb_W, c_t_1, xc_W, xc_b, w_x);
    k_lstm<<<BATCH, HID, 0, stream>>>(w_x, s_h, s_c, W_ih, W_hh, b_ih, b_hh,
                                      w_sth, out + OF_H, out + OF_C);
    k_decfea<<<BATCH, N2H, 0, stream>>>(w_sth, Ws_W, Ws_b, w_dec);
    k_cov_scores<<<NROWS / 8, 256, 0, stream>>>(enc_feat, w_dec, coverage, Wc_W, v_W, w_sc1);
    k_struct_scores_wmma<<<NROWS / 64, 256, 0, stream>>>(ri, Wr_W, w_dec, v_W, w_sc2);
    // path 1: coverage attention (renormalized), outputs c_t / attn_dist / coverage
    k_softmax_ctx<<<BATCH, 256, 0, stream>>>(w_sc1, mask, coverage, enc_out,
                                             w_att, out + OF_CT,
                                             out + OF_ATTN, out + OF_COV, 1);
    // path 2: structural attention (not renormalized), context only
    k_softmax_ctx<<<BATCH, 256, 0, stream>>>(w_sc2, mask, coverage, enc_out,
                                             w_ast, w_cst, nullptr, nullptr, 0);
    k_pgen<<<BATCH, 32, 0, stream>>>(w_cst, w_sth, w_x, pg_W, pg_b, w_pg, out + OF_PGEN);
    k_out1<<<BATCH, HID, 0, stream>>>(w_sth, w_cst, o1_W, o1_b, w_o1);
    k_logits_wmma<<<(VOCAB + 127) / 128, 256, 0, stream>>>(w_o1, o2_W, o2_b, w_log);
    k_vocab_softmax<<<BATCH, 256, 0, stream>>>(w_log, w_pg, out);
    k_scatter<<<(NROWS + 255) / 256, 256, 0, stream>>>(ext_voc, w_att, w_pg, out);
}
